// GroupedQueryAttention_77429670413055
// MI455X (gfx1250) — matrix operand order repaired and checked by reading, not compiled, not run on hardware
//
#include <hip/hip_runtime.h>
#include <hip/hip_bf16.h>
#include <stdint.h>

typedef __bf16 bf16_t;
typedef __attribute__((ext_vector_type(16))) __bf16 v16bf;
typedef __attribute__((ext_vector_type(8)))  __bf16 v8bf;
typedef __attribute__((ext_vector_type(8)))  float  v8f;

#define DEV __device__ __forceinline__

// D = A(16x32 bf16) * B(32x16 bf16) + C(16x16 f32)
DEV v8f wmma_bf16(v16bf a, v16bf b, v8f c) {
  return __builtin_amdgcn_wmma_f32_16x16x32_bf16(
      /*neg_a=*/false, a, /*neg_b=*/false, b,
      /*c_mod=*/(short)0, c, /*reuse_a=*/false, /*reuse_b=*/false);
}

// A-fragment: 16x32 bf16 from row-major source, leading dim lda.
// Lanes 0-15: row M=lane, K={0..7,16..23}; lanes 16-31: row M=lane-16, K={8..15,24..31}.
DEV v16bf load_a_frag(const bf16_t* __restrict__ A, int lda, int m0, int k0) {
  const int lane = threadIdx.x & 31;
  const int row  = m0 + (lane & 15);
  const int kb   = k0 + ((lane >> 4) << 3);
  const bf16_t* p = A + (size_t)row * lda + kb;
  v16bf a;
  ((v8bf*)&a)[0] = *(const v8bf*)(p);
  ((v8bf*)&a)[1] = *(const v8bf*)(p + 16);
  return a;
}

// B-fragment: 32x16 bf16 from column-major storage Bt[N][K] (ldb = K stride).
// Lanes 0-15: col N=lane, K={0..7,16..23}; lanes 16-31: col N=lane-16, K={8..15,24..31}.
DEV v16bf load_b_frag_t(const bf16_t* __restrict__ Bt, int ldb, int n0, int k0) {
  const int lane = threadIdx.x & 31;
  const int col  = n0 + (lane & 15);
  const int kb   = k0 + ((lane >> 4) << 3);
  const bf16_t* p = Bt + (size_t)col * ldb + kb;
  v16bf b;
  ((v8bf*)&b)[0] = *(const v8bf*)(p);
  ((v8bf*)&b)[1] = *(const v8bf*)(p + 16);
  return b;
}

// -------------------- elementwise converts --------------------

__global__ void __launch_bounds__(256) cvt_f32_bf16_kernel(
    const float* __restrict__ in, bf16_t* __restrict__ out, int n) {
  int i = blockIdx.x * 256 + threadIdx.x;
  if (i < n) out[i] = (bf16_t)in[i];
}

// W[K][N] f32 -> Wt[N][K] bf16 (writes contiguous)
__global__ void __launch_bounds__(256) transpose_cvt_kernel(
    const float* __restrict__ W, bf16_t* __restrict__ Wt, int K, int N) {
  int i = blockIdx.x * 256 + threadIdx.x;
  if (i >= K * N) return;
  int n = i / K, k = i - n * K;
  Wt[i] = (bf16_t)W[(size_t)k * N + n];
}

// -------------------- GEMM: out = A(MxK bf16) * Bt(NxK bf16)^T + bias --------------------
// Wave computes 32x64 (2 M-tiles x 4 N-tiles), double-buffered over k.
// MODE 0: bf16 row-major out; MODE 1: bf16 transposed out; MODE 2: f32 row-major out.
// Grid: (N/64, M/256), 256 threads (8 waves; wave w -> rows blockIdx.y*256 + w*32).

template<int MODE>
__global__ void __launch_bounds__(256) gemm_bias_kernel(
    const bf16_t* __restrict__ A, const bf16_t* __restrict__ Bt,
    const float* __restrict__ bias, void* __restrict__ out,
    int M, int N, int K, int ldo)
{
  const int wave = threadIdx.x >> 5;
  const int lane = threadIdx.x & 31;
  const int m0 = blockIdx.y * 256 + wave * 32;
  const int n0 = blockIdx.x * 64;

  v8f acc[2][4] = {};

  // prologue: fragments for k0 = 0
  v16bf a0 = load_a_frag(A, K, m0,      0);
  v16bf a1 = load_a_frag(A, K, m0 + 16, 0);
  v16bf b0 = load_b_frag_t(Bt, K, n0,      0);
  v16bf b1 = load_b_frag_t(Bt, K, n0 + 16, 0);
  v16bf b2 = load_b_frag_t(Bt, K, n0 + 32, 0);
  v16bf b3 = load_b_frag_t(Bt, K, n0 + 48, 0);

  for (int k0 = 0; k0 < K; k0 += 32) {
    // issue next-iteration loads first (redundant reload of k=0 on last iter; harmless)
    const int kn = (k0 + 32 < K) ? (k0 + 32) : 0;
    v16bf a0n = load_a_frag(A, K, m0,      kn);
    v16bf a1n = load_a_frag(A, K, m0 + 16, kn);
    v16bf b0n = load_b_frag_t(Bt, K, n0,      kn);
    v16bf b1n = load_b_frag_t(Bt, K, n0 + 16, kn);
    v16bf b2n = load_b_frag_t(Bt, K, n0 + 32, kn);
    v16bf b3n = load_b_frag_t(Bt, K, n0 + 48, kn);

    acc[0][0] = wmma_bf16(a0, b0, acc[0][0]);
    acc[0][1] = wmma_bf16(a0, b1, acc[0][1]);
    acc[0][2] = wmma_bf16(a0, b2, acc[0][2]);
    acc[0][3] = wmma_bf16(a0, b3, acc[0][3]);
    acc[1][0] = wmma_bf16(a1, b0, acc[1][0]);
    acc[1][1] = wmma_bf16(a1, b1, acc[1][1]);
    acc[1][2] = wmma_bf16(a1, b2, acc[1][2]);
    acc[1][3] = wmma_bf16(a1, b3, acc[1][3]);

    a0 = a0n; a1 = a1n; b0 = b0n; b1 = b1n; b2 = b2n; b3 = b3n;
  }

  const int hi = lane >> 4, nl = lane & 15;
#pragma unroll
  for (int mt = 0; mt < 2; ++mt) {
    const int mrow = m0 + mt * 16 + 8 * hi;
#pragma unroll
    for (int t = 0; t < 4; ++t) {
      const int n = n0 + t * 16 + nl;
      const float bv = bias[n];
#pragma unroll
      for (int r = 0; r < 8; ++r) acc[mt][t][r] += bv;

      if constexpr (MODE == 1) {
        v8bf pk;
#pragma unroll
        for (int r = 0; r < 8; ++r) pk[r] = (bf16_t)acc[mt][t][r];
        *(v8bf*)((bf16_t*)out + (size_t)n * ldo + mrow) = pk;
      } else if constexpr (MODE == 2) {
        float* o = (float*)out;
#pragma unroll
        for (int r = 0; r < 8; ++r)
          o[(size_t)(mrow + r) * ldo + n] = acc[mt][t][r];
      } else {
        bf16_t* o = (bf16_t*)out;
#pragma unroll
        for (int r = 0; r < 8; ++r)
          o[(size_t)(mrow + r) * ldo + n] = (bf16_t)acc[mt][t][r];
      }
    }
  }
}

// -------------------- Flash attention (GQA) --------------------
// One wave = 32 queries of one head. Workgroup = 8 waves all on the SAME kv
// group g (4 q-heads x 2 query blocks) so they stream identical K/V tiles
// in lockstep -> WGP$ sharing. Double-buffered K/V fragments over the key loop.
// Online softmax runs in the base-2 domain: S is pre-scaled by
// (1/sqrt(d))*log2(e) and p = exp2(s - m), so each exponential is a single
// v_exp_f32 with no extra multiply.
// Q:[4096][1024] bf16, K:[4096][256] bf16, Vt:[256][4096] bf16, ctx:[4096][1024] bf16.
__global__ void __launch_bounds__(256) flash_attn_kernel(
    const bf16_t* __restrict__ Qb, const bf16_t* __restrict__ Kb,
    const bf16_t* __restrict__ Vt, bf16_t* __restrict__ ctx)
{
  const int wave = threadIdx.x >> 5;
  const int lane = threadIdx.x & 31;
  const int hi = lane >> 4, nl = lane & 15;

  const int g  = blockIdx.x & 3;                 // kv group
  const int h  = g * 4 + (wave & 3);             // q head
  const int q0 = ((blockIdx.x >> 2) * 2 + (wave >> 2)) * 32;  // 32-query block

  const bf16_t* Qh  = Qb + (size_t)q0 * 1024 + h * 64;
  const bf16_t* Kh  = Kb + g * 64;                // stride 256
  const bf16_t* Vth = Vt + (size_t)g * 64 * 4096; // stride 4096

  const float SCALE2 = 0.125f * 1.44269504088896340736f; // (1/sqrt(64)) * log2(e)

  // Q^T B-fragments, reused for every key tile: [q-tile][d-chunk]
  v16bf bq[2][2];
#pragma unroll
  for (int qt = 0; qt < 2; ++qt) {
    bq[qt][0] = load_b_frag_t(Qh + (size_t)qt * 16 * 1024, 1024, 0, 0);
    bq[qt][1] = load_b_frag_t(Qh + (size_t)qt * 16 * 1024, 1024, 0, 32);
  }

  v8f o[2][4] = {};                              // O: [q-tile][d-tile] 16x16 each
  float m_run[2] = {-1e30f, -1e30f};             // running max (base-2 domain)
  float l_run[2] = {0.f, 0.f};

  // prologue fragments for kb0 = 0
  v16bf aK0 = load_a_frag(Kh, 256, 0, 0);
  v16bf aK1 = load_a_frag(Kh, 256, 0, 32);
  v16bf aK2 = load_a_frag(Kh, 256, 16, 0);
  v16bf aK3 = load_a_frag(Kh, 256, 16, 32);
  v16bf bV0 = load_b_frag_t(Vth, 4096, 0,  0);
  v16bf bV1 = load_b_frag_t(Vth, 4096, 16, 0);
  v16bf bV2 = load_b_frag_t(Vth, 4096, 32, 0);
  v16bf bV3 = load_b_frag_t(Vth, 4096, 48, 0);

  for (int kb0 = 0; kb0 < 4096; kb0 += 32) {
    // issue next K/V tile loads up front; consumed next iteration
    const int kn = (kb0 + 32 < 4096) ? (kb0 + 32) : 0;
    v16bf aK0n = load_a_frag(Kh, 256, kn,      0);
    v16bf aK1n = load_a_frag(Kh, 256, kn,      32);
    v16bf aK2n = load_a_frag(Kh, 256, kn + 16, 0);
    v16bf aK3n = load_a_frag(Kh, 256, kn + 16, 32);
    v16bf bV0n = load_b_frag_t(Vth, 4096, 0,  kn);
    v16bf bV1n = load_b_frag_t(Vth, 4096, 16, kn);
    v16bf bV2n = load_b_frag_t(Vth, 4096, 32, kn);
    v16bf bV3n = load_b_frag_t(Vth, 4096, 48, kn);

#pragma unroll
    for (int qt = 0; qt < 2; ++qt) {
      // S^T tiles (M=keys, N=queries), inner dim d=64
      v8f st0 = {}, st1 = {};
      st0 = wmma_bf16(aK0, bq[qt][0], st0);
      st0 = wmma_bf16(aK1, bq[qt][1], st0);
      st1 = wmma_bf16(aK2, bq[qt][0], st1);
      st1 = wmma_bf16(aK3, bq[qt][1], st1);

      // online softmax (base-2) for this lane's query column
      float tmax = -1e30f;
#pragma unroll
      for (int r = 0; r < 8; ++r) {
        st0[r] *= SCALE2;
        st1[r] *= SCALE2;
        tmax = fmaxf(tmax, fmaxf(st0[r], st1[r]));
      }
      tmax = fmaxf(tmax, __shfl_xor(tmax, 16, 32));
      const float mnew = fmaxf(m_run[qt], tmax);
      const float corr = __builtin_amdgcn_exp2f(m_run[qt] - mnew);

      float lsum = 0.f;
      v8bf pb0, pb1;
#pragma unroll
      for (int r = 0; r < 8; ++r) {
        const float p0 = __builtin_amdgcn_exp2f(st0[r] - mnew);
        const float p1 = __builtin_amdgcn_exp2f(st1[r] - mnew);
        lsum += p0 + p1;
        pb0[r] = (bf16_t)p0;
        pb1[r] = (bf16_t)p1;
      }
      lsum += __shfl_xor(lsum, 16, 32);
      l_run[qt] = l_run[qt] * corr + lsum;
      m_run[qt] = mnew;

      // S^T C-layout per lane == P A-fragment layout -> build aP in registers
      v16bf aP;
      ((v8bf*)&aP)[0] = pb0;
      ((v8bf*)&aP)[1] = pb1;

      // rescale O rows by this row's query correction (lane permute broadcast)
#pragma unroll
      for (int r = 0; r < 8; ++r) {
        const float s = __shfl(corr, 8 * hi + r, 32);
        o[qt][0][r] *= s; o[qt][1][r] *= s; o[qt][2][r] *= s; o[qt][3][r] *= s;
      }

      // O += P(16x32) * V(32x64)
      o[qt][0] = wmma_bf16(aP, bV0, o[qt][0]);
      o[qt][1] = wmma_bf16(aP, bV1, o[qt][1]);
      o[qt][2] = wmma_bf16(aP, bV2, o[qt][2]);
      o[qt][3] = wmma_bf16(aP, bV3, o[qt][3]);
    }

    aK0 = aK0n; aK1 = aK1n; aK2 = aK2n; aK3 = aK3n;
    bV0 = bV0n; bV1 = bV1n; bV2 = bV2n; bV3 = bV3n;
  }

  // normalize and store context[(q0+qt*16+m)][h*64 + d]
#pragma unroll
  for (int qt = 0; qt < 2; ++qt) {
    const float rl = 1.0f / l_run[qt];
#pragma unroll
    for (int r = 0; r < 8; ++r) {
      const float inv = __shfl(rl, 8 * hi + r, 32);
      const int m = q0 + qt * 16 + 8 * hi + r;
      bf16_t* co = ctx + (size_t)m * 1024 + h * 64 + nl;
      co[0]  = (bf16_t)(o[qt][0][r] * inv);
      co[16] = (bf16_t)(o[qt][1][r] * inv);
      co[32] = (bf16_t)(o[qt][2][r] * inv);
      co[48] = (bf16_t)(o[qt][3][r] * inv);
    }
  }
}

// -------------------- launch --------------------

extern "C" void kernel_launch(void* const* d_in, const int* in_sizes, int n_in,
                              void* d_out, int out_size, void* d_ws, size_t ws_size,
                              hipStream_t stream)
{
  const float* x  = (const float*)d_in[0];
  const float* Wq = (const float*)d_in[1];
  const float* bq = (const float*)d_in[2];
  const float* Wk = (const float*)d_in[3];
  const float* bk = (const float*)d_in[4];
  const float* Wv = (const float*)d_in[5];
  const float* bv = (const float*)d_in[6];
  const float* Wo = (const float*)d_in[7];
  const float* bo = (const float*)d_in[8];

  const int S = 4096, D = 1024, KV = 256;

  char* w = (char*)d_ws;
  size_t off = 0;
  auto alloc = [&](size_t bytes) -> void* {
    off = (off + 255) & ~(size_t)255;
    void* p = w + off;
    off += bytes;
    return p;
  };

  bf16_t* xb  = (bf16_t*)alloc((size_t)S * D  * sizeof(bf16_t));
  bf16_t* WqT = (bf16_t*)alloc((size_t)D * D  * sizeof(bf16_t));
  bf16_t* WkT = (bf16_t*)alloc((size_t)D * KV * sizeof(bf16_t));
  bf16_t* WvT = (bf16_t*)alloc((size_t)D * KV * sizeof(bf16_t));
  bf16_t* WoT = (bf16_t*)alloc((size_t)D * D  * sizeof(bf16_t));
  bf16_t* Qbuf= (bf16_t*)alloc((size_t)S * D  * sizeof(bf16_t));
  bf16_t* Kbuf= (bf16_t*)alloc((size_t)S * KV * sizeof(bf16_t));
  bf16_t* Vt  = (bf16_t*)alloc((size_t)KV * S * sizeof(bf16_t));
  bf16_t* ctx = (bf16_t*)alloc((size_t)S * D  * sizeof(bf16_t));
  (void)ws_size;

  const dim3 blk(256);

  cvt_f32_bf16_kernel<<<(S * D + 255) / 256, blk, 0, stream>>>(x, xb, S * D);
  transpose_cvt_kernel<<<(D * D  + 255) / 256, blk, 0, stream>>>(Wq, WqT, D, D);
  transpose_cvt_kernel<<<(D * KV + 255) / 256, blk, 0, stream>>>(Wk, WkT, D, KV);
  transpose_cvt_kernel<<<(D * KV + 255) / 256, blk, 0, stream>>>(Wv, WvT, D, KV);
  transpose_cvt_kernel<<<(D * D  + 255) / 256, blk, 0, stream>>>(Wo, WoT, D, D);

  // Q = xb @ Wq + bq -> bf16 [S][D]
  gemm_bias_kernel<0><<<dim3(D / 64, S / 256), blk, 0, stream>>>(xb, WqT, bq, Qbuf, S, D, D, D);
  // K = xb @ Wk + bk -> bf16 [S][KV]
  gemm_bias_kernel<0><<<dim3(KV / 64, S / 256), blk, 0, stream>>>(xb, WkT, bk, Kbuf, S, KV, D, KV);
  // V = xb @ Wv + bv -> bf16 transposed [KV][S]
  gemm_bias_kernel<1><<<dim3(KV / 64, S / 256), blk, 0, stream>>>(xb, WvT, bv, Vt, S, KV, D, S);

  // attention: 256 workgroups x 8 waves; workgroup = (kv group, pair of 32-query blocks)
  flash_attn_kernel<<<dim3(4 * (S / 64), 1), blk, 0, stream>>>(Qbuf, Kbuf, Vt, ctx);

  // out = ctx @ Wo + bo -> f32 [S][D] into d_out
  gemm_bias_kernel<2><<<dim3(D / 64, S / 256), blk, 0, stream>>>(ctx, WoT, bo, d_out, S, D, D, D);
}
